// DualGraphRelationModel_25683904430213
// MI455X (gfx1250) — matrix operand order repaired and checked by reading, not compiled, not run on hardware
//
#include <hip/hip_runtime.h>
#include <hip/hip_bf16.h>

// ---------------------------------------------------------------------------
// Types / WMMA helpers (CDNA5 gfx1250, wave32)
// ---------------------------------------------------------------------------
typedef __bf16 bf16;
typedef __attribute__((ext_vector_type(16))) __bf16 bf16x16;
typedef __attribute__((ext_vector_type(8)))  float  floatx8;

union FragAB { bf16x16 v; unsigned u[8]; };

__device__ __forceinline__ floatx8 fzero8() {
  floatx8 z = {0.f, 0.f, 0.f, 0.f, 0.f, 0.f, 0.f, 0.f};
  return z;
}

// A-matrix 16x32 bf16 fragment (M over lanes 0-15, K split per lane-half)
// VGPR i holds K = (i<4 ? 2i : 16+2(i-4)) + 8*(lane>>4) + {0,1}
__device__ __forceinline__ bf16x16 load_a_frag(const bf16* base, int rowStride, int lane) {
  FragAB r;
  int m = lane & 15;
  int khalf = lane >> 4;
  const bf16* row = base + (size_t)m * rowStride;
#pragma unroll
  for (int i = 0; i < 8; ++i) {
    int kb = ((i < 4) ? (i * 2) : (16 + (i - 4) * 2)) + khalf * 8;
    r.u[i] = *(const unsigned*)(row + kb);
  }
  return r.v;
}

// B-matrix 32x16 bf16 fragment, stored n-major (k contiguous):
// VGPR i holds K = (i<4 ? 2i : 16+2(i-4)) + 8*(lane>>4) + {0,1}, N = lane&15
__device__ __forceinline__ bf16x16 load_b_frag(const bf16* base, int rowStride, int lane) {
  FragAB r;
  int n = lane & 15;
  int kg = (lane >> 4) * 8;
  const bf16* row = base + (size_t)n * rowStride + kg;
#pragma unroll
  for (int i = 0; i < 8; ++i) {
    r.u[i] = *(const unsigned*)(row + ((i < 4) ? (i * 2) : (16 + (i - 4) * 2)));
  }
  return r.v;
}

__device__ __forceinline__ floatx8 wmma_bf16(bf16x16 a, bf16x16 b, floatx8 c) {
  return __builtin_amdgcn_wmma_f32_16x16x32_bf16(false, a, false, b, (short)0, c,
                                                 false, false);
}

__device__ __forceinline__ float sigf(float x) { return 1.f / (1.f + __expf(-x)); }

// ---------------------------------------------------------------------------
// CDNA5 async data movement (guarded; falls back to manual copies)
// ---------------------------------------------------------------------------
#if __has_builtin(__builtin_amdgcn_tensor_load_to_lds) && \
    __has_builtin(__builtin_amdgcn_s_wait_tensorcnt)
#define HAVE_TDM 1
#endif

#if __has_builtin(__builtin_amdgcn_global_load_async_to_lds_b128) && \
    __has_builtin(__builtin_amdgcn_s_wait_asynccnt)
#define HAVE_ASYNC_LDS 1
#endif

typedef unsigned int v4u __attribute__((ext_vector_type(4)));
typedef int          v8i __attribute__((ext_vector_type(8)));
typedef int          v4i __attribute__((ext_vector_type(4)));

#ifdef HAVE_TDM
// 2D bf16 tile load via Tensor Data Mover (D# per CDNA5 ISA).
// remK/remM: tensor extent remaining from tile start (OOB reads return zero).
// Optional LDS padding: (padAmount+1) dwords appended every (1<<(padInterval+1)) dwords.
__device__ __forceinline__ void tdm_load_2d_bf16(unsigned lds_byte_off,
                                                 const bf16* gsrc,
                                                 unsigned remK, unsigned remM,
                                                 unsigned lda,
                                                 unsigned tileK, unsigned tileM,
                                                 unsigned padInterval,
                                                 unsigned padAmount,
                                                 unsigned padEnable) {
  unsigned long long ga = (unsigned long long)gsrc;
  v4u g0;
  g0.x = 1u;                                              // count=1, user desc
  g0.y = lds_byte_off;                                    // lds_addr
  g0.z = (unsigned)(ga & 0xFFFFFFFFull);                  // global_addr[31:0]
  g0.w = (unsigned)((ga >> 32) & 0x01FFFFFFull) | 0x80000000u;  // addr[56:32]|type=2
  unsigned d0 = (1u << 16)                                // data_size = 2 bytes
              | (padEnable << 20)
              | (padInterval << 22) | (padAmount << 25);
  v8i g1;
  g1[0] = (int)d0;
  g1[1] = (int)((remK & 0xFFFFu) << 16);                  // tensor_dim0 lo16
  g1[2] = (int)(((remK >> 16) & 0xFFFFu) | ((remM & 0xFFFFu) << 16));
  g1[3] = (int)(((remM >> 16) & 0xFFFFu) | ((tileK & 0xFFFFu) << 16)); // tile_dim0
  g1[4] = (int)(tileM & 0xFFFFu);                         // tile_dim1, tile_dim2=0
  g1[5] = (int)lda;                                       // tensor_dim0_stride lo32
  g1[6] = 0;
  g1[7] = 0;
  v4i z4 = {0, 0, 0, 0};
#if __clang_major__ >= 23
  v8i z8 = {0, 0, 0, 0, 0, 0, 0, 0};
  __builtin_amdgcn_tensor_load_to_lds(g0, g1, z4, z4, z8, 0);
#else
  __builtin_amdgcn_tensor_load_to_lds(g0, g1, z4, z4, 0);
#endif
}
#endif  // HAVE_TDM

#ifdef HAVE_ASYNC_LDS
typedef __attribute__((address_space(1))) v4i gv4i;   // global int4
typedef __attribute__((address_space(3))) v4i lv4i;   // LDS int4
__device__ __forceinline__ void async_copy_b128(const void* gsrc, void* ldst) {
  __builtin_amdgcn_global_load_async_to_lds_b128(
      (gv4i*)(unsigned long long)gsrc,
      (lv4i*)(unsigned)(unsigned long long)ldst, 0, 0);
}
#endif

// ---------------------------------------------------------------------------
// Utility kernels
// ---------------------------------------------------------------------------
__global__ void zero_f32_kernel(float* p, long long n) {
  long long i = (long long)blockIdx.x * blockDim.x + threadIdx.x;
  if (i < n) p[i] = 0.f;
}

__global__ void cvt_bf16_kernel(bf16* dst, const float* src, long long n) {
  long long i = (long long)blockIdx.x * blockDim.x + threadIdx.x;
  if (i < n) dst[i] = (bf16)src[i];
}

// dst[k*ld + n] = src[n*K + k]   (transpose + convert); src is [Nrows, Kcols]
__global__ void cvtT_bf16_kernel(bf16* dst, int ld, const float* src,
                                 int Nrows, int Kcols) {
  long long i = (long long)blockIdx.x * blockDim.x + threadIdx.x;
  long long tot = (long long)Nrows * Kcols;
  if (i < tot) {
    int k = (int)(i % Kcols);
    int n = (int)(i / Kcols);
    dst[(size_t)k * ld + n] = (bf16)src[(size_t)n * Kcols + k];
  }
}

// ---------------------------------------------------------------------------
// Normalized-adjacency build (PyG GCNConv norm): A[dst,src] += d^-1/2 * d^-1/2
// ---------------------------------------------------------------------------
__global__ void deg_kernel(const int* ei, int E, float* deg, int n) {
  int t = blockIdx.x * blockDim.x + threadIdx.x;
  if (t < E)            atomicAdd(&deg[ei[E + t]], 1.f);   // dst degree
  else if (t < E + n)   atomicAdd(&deg[t - E], 1.f);       // self loop
}

__global__ void adj_kernel(const int* ei, int E, const float* deg, float* A, int n) {
  int t = blockIdx.x * blockDim.x + threadIdx.x;
  if (t < E) {
    int s = ei[t], d = ei[E + t];
    float v = rsqrtf(deg[d]) * rsqrtf(deg[s]);
    atomicAdd(&A[(size_t)d * n + s], v);
  } else if (t < E + n) {
    int i = t - E;
    atomicAdd(&A[(size_t)i * n + i], 1.f / deg[i]);
  }
}

// ---------------------------------------------------------------------------
// General bf16 WMMA GEMM:  C = act( A(MxK) * B(KxN) + bias + resid )
// Workgroup: 256 threads (8 waves) -> 64x128 tile; wave -> 16x64 (1x4 tiles).
// A tile staged by TDM DMA (wave 0) when available; B staged manually with
// a transpose into n-major LDS (branchless clamped loads at the edges).
// ---------------------------------------------------------------------------
#define GTM 64
#define GTN 128
#define GTK 32
#define GPAD 4

__global__ __launch_bounds__(256)
void gemm_bf16_kernel(const bf16* __restrict__ A, long long strideA, int lda,
                      const bf16* __restrict__ Bm, long long strideB, int ldb,
                      const float* __restrict__ bias,
                      const float* __restrict__ resid, long long strideR, int ldr,
                      float* __restrict__ C, long long strideC, int ldc,
                      int M, int N, int K, int relu)
{
  __shared__ bf16 As[GTM][GTK + GPAD];   // 36 bf16 = 18 dwords per row
  __shared__ bf16 Bs[GTN][GTK + GPAD];

  int bz = blockIdx.z;
  A  += (long long)bz * strideA;
  Bm += (long long)bz * strideB;
  C  += (long long)bz * strideC;
  const float* R = resid ? resid + (long long)bz * strideR : nullptr;

  int n0 = blockIdx.x * GTN;
  int m0 = blockIdx.y * GTM;
  int tid = threadIdx.x, lane = tid & 31, w = tid >> 5;
  int mt  = w & 3;             // 4 m-subtiles of 16
  int ntb = (w >> 2) * 4;      // 4 n-subtiles per wave

  floatx8 acc[4];
#pragma unroll
  for (int j = 0; j < 4; ++j) acc[j] = fzero8();

  for (int kk = 0; kk < K; kk += GTK) {
#ifdef HAVE_TDM
    // A tile (64x32 bf16) via Tensor Data Mover, wave 0 only (EXEC-independent,
    // TENSORcnt-tracked). Pad 2 dwords every 16 dwords -> LDS row stride 36 bf16.
    if (tid < 32) {
      tdm_load_2d_bf16((unsigned)(unsigned long long)(void*)&As[0][0],
                       A + (size_t)m0 * lda + kk,
                       (unsigned)(K - kk), (unsigned)(M - m0),
                       (unsigned)lda, GTK, GTM, /*padInterval=*/3,
                       /*padAmount=*/1, /*padEnable=*/1);
    }
#else
    for (int idx = tid; idx < GTM * GTK; idx += 256) {
      int kc = idx & 31, m = idx >> 5;
      int gm = m0 + m, gk = kk + kc;
      int cm = gm < M ? gm : M - 1;
      int ck = gk < K ? gk : K - 1;
      bf16 v = A[(size_t)cm * lda + ck];
      As[m][kc] = (gm < M && gk < K) ? v : (bf16)0.0f;
    }
#endif
    // B tile (32x128), transposed into n-major LDS; branchless clamped loads
    for (int idx = tid; idx < GTN * GTK; idx += 256) {
      int n = idx & 127, kc = idx >> 7;
      int gn = n0 + n, gk = kk + kc;
      int cn = gn < N ? gn : N - 1;
      int ck = gk < K ? gk : K - 1;
      bf16 v = Bm[(size_t)ck * ldb + cn];
      Bs[n][kc] = (gn < N && gk < K) ? v : (bf16)0.0f;
    }
#ifdef HAVE_TDM
    if (tid < 32) __builtin_amdgcn_s_wait_tensorcnt(0);
#endif
    __syncthreads();

    bf16x16 af = load_a_frag(&As[mt * 16][0], GTK + GPAD, lane);
#pragma unroll
    for (int j = 0; j < 4; ++j) {
      bf16x16 bfr = load_b_frag(&Bs[(ntb + j) * 16][0], GTK + GPAD, lane);
      acc[j] = wmma_bf16(af, bfr, acc[j]);
    }
    __syncthreads();
  }

  // epilogue: C-frag element r -> M = r + 8*(lane>>4), N = lane&15
  int ncol = lane & 15;
  int mhi  = (lane >> 4) * 8;
#pragma unroll
  for (int j = 0; j < 4; ++j) {
    int gn = n0 + (ntb + j) * 16 + ncol;
#pragma unroll
    for (int r = 0; r < 8; ++r) {
      int gm = m0 + mt * 16 + mhi + r;
      if (gm < M && gn < N) {
        float v = acc[j][r];
        if (bias)  v += bias[gn];
        if (R)     v += R[(size_t)gm * ldr + gn];
        if (relu)  v = fmaxf(v, 0.f);
        C[(size_t)gm * ldc + gn] = v;
      }
    }
  }
}

// ---------------------------------------------------------------------------
// Persistent BiLSTM scan. One block per direction, 16 waves.
// G[32,1024] = h_bf16 @ Whh^T (WMMA) + XW[t] ; gates in-register.
// ---------------------------------------------------------------------------
__global__ __launch_bounds__(512)
void lstm_scan_kernel(const float* __restrict__ XW,    // [32][300][2048] (bias folded)
                      const bf16*  __restrict__ Whh,   // [2][1024][256]  n-major bf16
                      float* __restrict__ gbuf_all,    // [2][32][1024] scratch
                      float* __restrict__ text_f32,    // [32][300][512]
                      bf16*  __restrict__ text_bf)     // [32][300][512]
{
  const int S = 300, HL = 256;
  int dir = blockIdx.x;
  const bf16* W = Whh + (size_t)dir * 1024 * HL;
  float* gbuf = gbuf_all + (size_t)dir * 32 * 1024;

  __shared__ bf16 hbf[32][264];

  int tid = threadIdx.x, lane = tid & 31, w = tid >> 5;
  int mt  = w & 1;            // 2 m-tiles (M=32)
  int ntb = (w >> 1) * 8;     // 8 n-tiles per wave (N=1024 -> 64 tiles)

  for (int idx = tid; idx < 32 * 264; idx += 512) ((bf16*)hbf)[idx] = (bf16)0.0f;

  float creg[16];
#pragma unroll
  for (int q = 0; q < 16; ++q) creg[q] = 0.f;
  int b  = tid >> 4;          // 0..31
  int jb = (tid & 15) * 16;   // owns h/c cells [b][jb..jb+15]
  __syncthreads();

  for (int t = 0; t < S; ++t) {
    int xs = dir ? (S - 1 - t) : t;

    // --- G = h @ Whh^T via WMMA (K=256 -> 8 steps) ---
    floatx8 acc[8];
#pragma unroll
    for (int j = 0; j < 8; ++j) acc[j] = fzero8();

    for (int kk = 0; kk < HL; kk += 32) {
      bf16x16 af = load_a_frag(&hbf[mt * 16][kk], 264, lane);
#pragma unroll
      for (int j = 0; j < 8; ++j) {
        bf16x16 bfr = load_b_frag(W + (size_t)((ntb + j) * 16) * HL + kk, HL, lane);
        acc[j] = wmma_bf16(af, bfr, acc[j]);
      }
    }

    int ncol = lane & 15;
    int mhi  = (lane >> 4) * 8;
#pragma unroll
    for (int j = 0; j < 8; ++j) {
      int gn = (ntb + j) * 16 + ncol;
#pragma unroll
      for (int r = 0; r < 8; ++r) {
        int gm = mt * 16 + mhi + r;  // batch index
        gbuf[(size_t)gm * 1024 + gn] =
            acc[j][r] + XW[((size_t)gm * S + xs) * 2048 + (size_t)dir * 1024 + gn];
      }
    }
    __threadfence();
    __syncthreads();

    // --- gates (thread owns 16 cells) ---
#pragma unroll
    for (int q = 0; q < 16; ++q) {
      int jj = jb + q;
      float ig = gbuf[(size_t)b * 1024 + jj];
      float fg = gbuf[(size_t)b * 1024 + 256 + jj];
      float gg = gbuf[(size_t)b * 1024 + 512 + jj];
      float og = gbuf[(size_t)b * 1024 + 768 + jj];
      float c  = sigf(fg) * creg[q] + sigf(ig) * tanhf(gg);
      float h  = sigf(og) * tanhf(c);
      creg[q]  = c;
      size_t o = ((size_t)b * S + xs) * 512 + (size_t)dir * 256 + jj;
      text_f32[o] = h;
      text_bf[o]  = (bf16)h;
      hbf[b][jj]  = (bf16)h;
    }
    __syncthreads();
  }
}

// ---------------------------------------------------------------------------
// Per-(node, head) attention: scores, softmax, A@V  (tiny FLOPs, f32 VALU)
// q=gcn_feat[L=32 over batch], k=v=text. Layout rows = b*300+s.
// ---------------------------------------------------------------------------
__global__ __launch_bounds__(256)
void attn_kernel(const float* __restrict__ qh,   // [9600][512]
                 const float* __restrict__ kv,   // [9600][1024] (k|v)
                 float* __restrict__ ctx)        // [9600][512]
{
  const int Sn = 300;
  int n = blockIdx.x, h = blockIdx.y;
  __shared__ float qs[32][64], ks[32][64], vs[32][64], sc[32][36];
  int tid = threadIdx.x;

  for (int idx = tid; idx < 32 * 64; idx += 256) {
    int l = idx >> 6, d = idx & 63;
    size_t row = (size_t)l * Sn + n;
    qs[l][d] = qh[row * 512 + h * 64 + d];
    ks[l][d] = kv[row * 1024 + h * 64 + d];
    vs[l][d] = kv[row * 1024 + 512 + h * 64 + d];
  }
  __syncthreads();

  for (int idx = tid; idx < 32 * 32; idx += 256) {
    int l = idx >> 5, m = idx & 31;
    float s = 0.f;
#pragma unroll
    for (int d = 0; d < 64; ++d) s += qs[l][d] * ks[m][d];
    sc[l][m] = s * 0.125f;  // 1/sqrt(64)
  }
  __syncthreads();

  if (tid < 32) {
    float mx = -3.4e38f;
    for (int m = 0; m < 32; ++m) mx = fmaxf(mx, sc[tid][m]);
    float sum = 0.f;
    for (int m = 0; m < 32; ++m) { float e = __expf(sc[tid][m] - mx); sc[tid][m] = e; sum += e; }
    float inv = 1.f / sum;
    for (int m = 0; m < 32; ++m) sc[tid][m] *= inv;
  }
  __syncthreads();

  for (int idx = tid; idx < 32 * 64; idx += 256) {
    int l = idx >> 6, d = idx & 63;
    float s = 0.f;
#pragma unroll
    for (int m = 0; m < 32; ++m) s += sc[l][m] * vs[m][d];
    ctx[((size_t)l * Sn + n) * 512 + h * 64 + d] = s;
  }
}

// ---------------------------------------------------------------------------
// Final MLP: out[32,256] = combined[32,153600] @ mlp_w[153600,256] + b
// Bandwidth-bound (157MB of f32 weights @ 23.3 TB/s). Coalesced weight
// stream; activations async-DMA'd into LDS and reused across all 32 rows.
// ---------------------------------------------------------------------------
#define MLP_BLOCKS 32
#define MLP_CHUNK  480

__global__ __launch_bounds__(256)
void mlp_partial_kernel(const float* __restrict__ X,    // [32][153600]
                        const float* __restrict__ Wt,   // [153600][256]
                        float* __restrict__ part)       // [MLP_BLOCKS][32][256]
{
  __shared__ __align__(16) float xs[32][MLP_CHUNK];
  const long long KTOT = 153600;
  const long long perBlk = KTOT / MLP_BLOCKS;  // 4800
  long long i0 = (long long)blockIdx.x * perBlk;
  int j = threadIdx.x;

  float acc[32];
#pragma unroll
  for (int bb = 0; bb < 32; ++bb) acc[bb] = 0.f;

  for (long long c = 0; c < perBlk; c += MLP_CHUNK) {
#ifdef HAVE_ASYNC_LDS
    // 16B chunks: 32 rows x 120 chunks = 3840 -> 15 per thread, ASYNCcnt-tracked
    for (int idx = threadIdx.x; idx < 32 * (MLP_CHUNK / 4); idx += 256) {
      int bb = idx / (MLP_CHUNK / 4), c4 = idx % (MLP_CHUNK / 4);
      async_copy_b128(&X[(size_t)bb * KTOT + i0 + c + c4 * 4], &xs[bb][c4 * 4]);
    }
    __builtin_amdgcn_s_wait_asynccnt(0);
#else
    for (int idx = threadIdx.x; idx < 32 * MLP_CHUNK; idx += 256) {
      int bb = idx / MLP_CHUNK, ii = idx % MLP_CHUNK;
      xs[bb][ii] = X[(size_t)bb * KTOT + i0 + c + ii];
    }
#endif
    __syncthreads();
    for (int ii = 0; ii < MLP_CHUNK; ++ii) {
      __builtin_prefetch(&Wt[(size_t)(i0 + c + ii + 64) * 256 + j], 0, 0);
      float wv = Wt[(size_t)(i0 + c + ii) * 256 + j];
#pragma unroll
      for (int bb = 0; bb < 32; ++bb) acc[bb] += xs[bb][ii] * wv;
    }
    __syncthreads();
  }
#pragma unroll
  for (int bb = 0; bb < 32; ++bb)
    part[((size_t)blockIdx.x * 32 + bb) * 256 + j] = acc[bb];
}

__global__ void mlp_reduce_kernel(const float* __restrict__ part,
                                  const float* __restrict__ bias,
                                  float* __restrict__ out) {
  int idx = blockIdx.x * blockDim.x + threadIdx.x;  // 8192
  if (idx < 32 * 256) {
    float s = bias[idx & 255];
    for (int p = 0; p < MLP_BLOCKS; ++p) s += part[(size_t)p * 8192 + idx];
    out[idx] = s;
  }
}

// ---------------------------------------------------------------------------
// Host orchestration
// ---------------------------------------------------------------------------
static inline void launch_gemm(hipStream_t st,
                               const bf16* A, long long sA, int lda,
                               const bf16* B, long long sB, int ldb,
                               const float* bias,
                               const float* resid, long long sR, int ldr,
                               float* C, long long sC, int ldc,
                               int M, int N, int K, int relu, int batch) {
  dim3 grid((N + GTN - 1) / GTN, (M + GTM - 1) / GTM, batch);
  gemm_bf16_kernel<<<grid, 256, 0, st>>>(A, sA, lda, B, sB, ldb, bias,
                                         resid, sR, ldr, C, sC, ldc, M, N, K, relu);
}

static inline void launch_cvt(hipStream_t st, bf16* dst, const float* src, long long n) {
  cvt_bf16_kernel<<<dim3((unsigned)((n + 255) / 256)), 256, 0, st>>>(dst, src, n);
}

extern "C" void kernel_launch(void* const* d_in, const int* in_sizes, int n_in,
                              void* d_out, int out_size, void* d_ws, size_t ws_size,
                              hipStream_t stream) {
  (void)n_in; (void)out_size; (void)ws_size;
  const int B = 32, S = 300, DBERT = 768, HL = 256, DM = 512, GH = 128;
  const int BS = B * S;  // 9600

  // --- inputs ---
  const float* bert = (const float*)d_in[0];
  const int*   ei_i = (const int*)d_in[1];
  const int*   ei_s = (const int*)d_in[2];
  int E_i = in_sizes[1] / 2, E_s = in_sizes[2] / 2;
  const float* w_ih_f = (const float*)d_in[3];
  const float* w_hh_f = (const float*)d_in[4];
  const float* b_f    = (const float*)d_in[5];
  const float* w_ih_b = (const float*)d_in[6];
  const float* w_hh_b = (const float*)d_in[7];
  const float* b_b    = (const float*)d_in[8];
  const float* gi_w1 = (const float*)d_in[9];   const float* gi_b1 = (const float*)d_in[10];
  const float* gi_w2 = (const float*)d_in[11];  const float* gi_b2 = (const float*)d_in[12];
  const float* gs_w1 = (const float*)d_in[13];  const float* gs_b1 = (const float*)d_in[14];
  const float* gs_w2 = (const float*)d_in[15];  const float* gs_b2 = (const float*)d_in[16];
  const float* fc_w  = (const float*)d_in[17];  const float* fc_b  = (const float*)d_in[18];
  const float* proj_w = (const float*)d_in[19]; const float* proj_b = (const float*)d_in[20];
  const float* attn_in_w  = (const float*)d_in[21]; const float* attn_in_b  = (const float*)d_in[22];
  const float* attn_out_w = (const float*)d_in[23]; const float* attn_out_b = (const float*)d_in[24];
  const float* mlp_w = (const float*)d_in[25];  const float* mlp_b = (const float*)d_in[26];
  float* out = (float*)d_out;

  // --- workspace carve-up ---
  char* cur = (char*)d_ws;
  auto alloc = [&](size_t bytes) -> void* {
    void* r = (void*)cur;
    cur += (bytes + 255) & ~(size_t)255;
    return r;
  };
  bf16*  bert_bf   = (bf16*)alloc((size_t)BS * DBERT * 2);
  bf16*  wih_pack  = (bf16*)alloc((size_t)DBERT * 2048 * 2);
  bf16*  whh_pack  = (bf16*)alloc((size_t)2 * 1024 * HL * 2);
  float* bias2048  = (float*)alloc(2048 * 4);
  float* XW        = (float*)alloc((size_t)BS * 2048 * 4);
  float* gbuf      = (float*)alloc((size_t)2 * 32 * 1024 * 4);
  float* text_f32  = (float*)alloc((size_t)BS * DM * 4);
  bf16*  text_bf   = (bf16*)alloc((size_t)BS * DM * 2);
  float* deg       = (float*)alloc((size_t)2 * S * 4);
  float* Afull     = (float*)alloc((size_t)2 * S * S * 4);
  bf16*  Abf       = (bf16*)alloc((size_t)2 * S * S * 2);
  float* z_f32     = (float*)alloc((size_t)BS * DM * 4);
  bf16*  z_bf      = (bf16*)alloc((size_t)BS * DM * 2);
  float* x1_f32    = (float*)alloc((size_t)BS * GH * 4);
  bf16*  x1_bf     = (bf16*)alloc((size_t)BS * GH * 2);
  float* cat_f32   = (float*)alloc((size_t)BS * 256 * 4);
  bf16*  cat_bf    = (bf16*)alloc((size_t)BS * 256 * 2);
  bf16*  gi_w1_bf  = (bf16*)alloc((size_t)DM * GH * 2);
  bf16*  gi_w2_bf  = (bf16*)alloc((size_t)GH * GH * 2);
  bf16*  gs_w1_bf  = (bf16*)alloc((size_t)DM * GH * 2);
  bf16*  gs_w2_bf  = (bf16*)alloc((size_t)GH * GH * 2);
  bf16*  fc_w_bf   = (bf16*)alloc((size_t)256 * 256 * 2);
  bf16*  proj_w_bf = (bf16*)alloc((size_t)256 * DM * 2);
  bf16*  ain_wT    = (bf16*)alloc((size_t)DM * 1536 * 2);
  bf16*  aout_wT   = (bf16*)alloc((size_t)DM * DM * 2);
  float* gcn_feat  = (float*)alloc((size_t)BS * DM * 4);
  bf16*  gcn_bf    = (bf16*)alloc((size_t)BS * DM * 2);
  float* qh        = (float*)alloc((size_t)BS * DM * 4);
  float* kv        = (float*)alloc((size_t)BS * 1024 * 4);
  float* ctx       = (float*)alloc((size_t)BS * DM * 4);
  bf16*  ctx_bf    = (bf16*)alloc((size_t)BS * DM * 2);
  float* combined  = (float*)alloc((size_t)BS * DM * 4);
  float* part      = (float*)alloc((size_t)MLP_BLOCKS * 32 * 256 * 4);

  // --- 0. zero adjacency state ---
  {
    long long nz = 2 * S + 2 * S * S;
    zero_f32_kernel<<<dim3((unsigned)((nz + 255) / 256)), 256, 0, stream>>>(deg, nz);
  }

  // --- 1. pack inputs/weights to bf16 ---
  launch_cvt(stream, bert_bf, bert, (long long)BS * DBERT);
  {
    long long n = (long long)1024 * DBERT;
    cvtT_bf16_kernel<<<dim3((unsigned)((n + 255) / 256)), 256, 0, stream>>>(
        wih_pack, 2048, w_ih_f, 1024, DBERT);
    cvtT_bf16_kernel<<<dim3((unsigned)((n + 255) / 256)), 256, 0, stream>>>(
        wih_pack + 1024, 2048, w_ih_b, 1024, DBERT);
  }
  launch_cvt(stream, whh_pack, w_hh_f, (long long)1024 * HL);
  launch_cvt(stream, whh_pack + (size_t)1024 * HL, w_hh_b, (long long)1024 * HL);
  (void)hipMemcpyAsync(bias2048, b_f, 1024 * 4, hipMemcpyDeviceToDevice, stream);
  (void)hipMemcpyAsync(bias2048 + 1024, b_b, 1024 * 4, hipMemcpyDeviceToDevice, stream);

  // --- 2. XW = bert @ [Wih_f^T | Wih_b^T] + [b_f|b_b]  (30 GFLOP, WMMA) ---
  launch_gemm(stream, bert_bf, 0, DBERT, wih_pack, 0, 2048, bias2048,
              nullptr, 0, 0, XW, 0, 2048, BS, 2048, DBERT, 0, 1);

  // --- 3. BiLSTM scan (persistent, 1 block/direction) ---
  lstm_scan_kernel<<<2, 512, 0, stream>>>(XW, whh_pack, gbuf, text_f32, text_bf);

  // --- 4. adjacency matrices ---
  deg_kernel<<<1, 256, 0, stream>>>(ei_i, E_i, deg, S);
  deg_kernel<<<1, 256, 0, stream>>>(ei_s, E_s, deg + S, S);
  adj_kernel<<<2, 256, 0, stream>>>(ei_i, E_i, deg, Afull, S);
  adj_kernel<<<2, 256, 0, stream>>>(ei_s, E_s, deg + S, Afull + (size_t)S * S, S);
  launch_cvt(stream, Abf, Afull, (long long)2 * S * S);
  bf16* Ai_bf = Abf;
  bf16* As_bf = Abf + (size_t)S * S;

  // --- 5. GCN weight packs ---
  launch_cvt(stream, gi_w1_bf, gi_w1, (long long)DM * GH);
  launch_cvt(stream, gi_w2_bf, gi_w2, (long long)GH * GH);
  launch_cvt(stream, gs_w1_bf, gs_w1, (long long)DM * GH);
  launch_cvt(stream, gs_w2_bf, gs_w2, (long long)GH * GH);
  launch_cvt(stream, fc_w_bf, fc_w, (long long)256 * 256);
  launch_cvt(stream, proj_w_bf, proj_w, (long long)256 * DM);
  {
    long long n1 = (long long)1536 * DM;
    cvtT_bf16_kernel<<<dim3((unsigned)((n1 + 255) / 256)), 256, 0, stream>>>(
        ain_wT, 1536, attn_in_w, 1536, DM);
    long long n2 = (long long)DM * DM;
    cvtT_bf16_kernel<<<dim3((unsigned)((n2 + 255) / 256)), 256, 0, stream>>>(
        aout_wT, DM, attn_out_w, DM, DM);
  }

  // --- 6. dual GCN (batched WMMA aggregations + projections) ---
  launch_gemm(stream, Ai_bf, 0, S, text_bf, (long long)S * DM, DM, nullptr,
              nullptr, 0, 0, z_f32, (long long)S * DM, DM, S, DM, S, 0, B);
  launch_cvt(stream, z_bf, z_f32, (long long)BS * DM);
  launch_gemm(stream, z_bf, 0, DM, gi_w1_bf, 0, GH, gi_b1,
              nullptr, 0, 0, x1_f32, 0, GH, BS, GH, DM, 1, 1);
  launch_cvt(stream, x1_bf, x1_f32, (long long)BS * GH);
  launch_gemm(stream, Ai_bf, 0, S, x1_bf, (long long)S * GH, GH, nullptr,
              nullptr, 0, 0, z_f32, (long long)S * GH, GH, S, GH, S, 0, B);
  launch_cvt(stream, z_bf, z_f32, (long long)BS * GH);
  launch_gemm(stream, z_bf, 0, GH, gi_w2_bf, 0, GH, gi_b2,
              nullptr, 0, 0, cat_f32, 0, 256, BS, GH, GH, 1, 1);
  launch_gemm(stream, As_bf, 0, S, text_bf, (long long)S * DM, DM, nullptr,
              nullptr, 0, 0, z_f32, (long long)S * DM, DM, S, DM, S, 0, B);
  launch_cvt(stream, z_bf, z_f32, (long long)BS * DM);
  launch_gemm(stream, z_bf, 0, DM, gs_w1_bf, 0, GH, gs_b1,
              nullptr, 0, 0, x1_f32, 0, GH, BS, GH, DM, 1, 1);
  launch_cvt(stream, x1_bf, x1_f32, (long long)BS * GH);
  launch_gemm(stream, As_bf, 0, S, x1_bf, (long long)S * GH, GH, nullptr,
              nullptr, 0, 0, z_f32, (long long)S * GH, GH, S, GH, S, 0, B);
  launch_cvt(stream, z_bf, z_f32, (long long)BS * GH);
  launch_gemm(stream, z_bf, 0, GH, gs_w2_bf, 0, GH, gs_b2,
              nullptr, 0, 0, cat_f32 + 128, 0, 256, BS, GH, GH, 1, 1);

  // g = cat @ fc_w + fc_b ; gcn_feat = g @ proj_w + proj_b
  launch_cvt(stream, cat_bf, cat_f32, (long long)BS * 256);
  launch_gemm(stream, cat_bf, 0, 256, fc_w_bf, 0, 256, fc_b,
              nullptr, 0, 0, z_f32, 0, 256, BS, 256, 256, 0, 1);
  launch_cvt(stream, z_bf, z_f32, (long long)BS * 256);
  launch_gemm(stream, z_bf, 0, 256, proj_w_bf, 0, DM, proj_b,
              nullptr, 0, 0, gcn_feat, 0, DM, BS, DM, 256, 0, 1);
  launch_cvt(stream, gcn_bf, gcn_feat, (long long)BS * DM);

  // --- 7. MHA over batch axis ---
  launch_gemm(stream, gcn_bf, 0, DM, ain_wT, 0, 1536, attn_in_b,
              nullptr, 0, 0, qh, 0, DM, BS, DM, DM, 0, 1);
  launch_gemm(stream, text_bf, 0, DM, ain_wT + 512, 0, 1536, attn_in_b + 512,
              nullptr, 0, 0, kv, 0, 1024, BS, 1024, DM, 0, 1);
  attn_kernel<<<dim3(S, 8), 256, 0, stream>>>(qh, kv, ctx);
  launch_cvt(stream, ctx_bf, ctx, (long long)BS * DM);
  launch_gemm(stream, ctx_bf, 0, DM, aout_wT, 0, DM, attn_out_b,
              gcn_feat, 0, DM, combined, 0, DM, BS, DM, DM, 0, 1);

  // --- 8. final MLP (bandwidth-bound, f32 streaming) ---
  mlp_partial_kernel<<<MLP_BLOCKS, 256, 0, stream>>>(combined, mlp_w, part);
  mlp_reduce_kernel<<<32, 256, 0, stream>>>(part, mlp_b, out);
}
